// RegDGCNN_19456201851581
// MI455X (gfx1250) — matrix operand order repaired and checked by reading, not compiled, not run on hardware
//
#include <hip/hip_runtime.h>
#include <cstdint>
#include <cstddef>

typedef __attribute__((ext_vector_type(16))) _Float16 v16h;
typedef __attribute__((ext_vector_type(8)))  _Float16 v8h;
typedef __attribute__((ext_vector_type(8)))  float    v8f;

#define BNS 0.9999950000374998f  /* 1/sqrt(1+1e-5) */
#define NPTS 1024
#define BATCH 8
#define KNN_K 20

// ---------------------------------------------------------------------------
// concat [pos|normals] -> x0 [B,N,6]
// ---------------------------------------------------------------------------
__global__ void concat_kernel(const float* __restrict__ pos,
                              const float* __restrict__ nrm,
                              float* __restrict__ x0) {
  int i = blockIdx.x * blockDim.x + threadIdx.x;
  const int total = BATCH * NPTS * 6;
  if (i >= total) return;
  int pn = i / 6, c = i - pn * 6;
  x0[i] = (c < 3) ? pos[pn * 3 + c] : nrm[pn * 3 + (c - 3)];
}

// ---------------------------------------------------------------------------
// Swizzle a weight matrix W[K,C] (f32) into f16 "B-fragment-ready" layout:
// element (k,c) -> out[((k/32)*C + c)*32 + ((k>>4)&1)*16 + (k&15)]
// so one lane's 16-half fragment (16 consecutive k at fixed c) is contiguous.
// Rows k >= K are zero-padded up to KP.
// ---------------------------------------------------------------------------
__global__ void wswizzle_kernel(const float* __restrict__ W, _Float16* __restrict__ out,
                                int K, int KP, int C) {
  int i = blockIdx.x * blockDim.x + threadIdx.x;
  if (i >= KP * C) return;
  int k = i / C, c = i - k * C;
  float v = (k < K) ? W[(size_t)k * C + c] : 0.f;
  out[(((size_t)(k >> 5) * C + c) << 5) + (((k >> 4) & 1) << 4) + (k & 15)] = (_Float16)v;
}

// ---------------------------------------------------------------------------
// squared norms per point
// ---------------------------------------------------------------------------
template <int F>
__global__ void sqnorm_kernel(const float* __restrict__ x, float* __restrict__ sq) {
  int i = blockIdx.x * blockDim.x + threadIdx.x;
  if (i >= BATCH * NPTS) return;
  const float* p = x + (size_t)i * F;
  float s = 0.f;
#pragma unroll 4
  for (int f = 0; f < F; ++f) s += p[f] * p[f];
  sq[i] = s;
}

// ---------------------------------------------------------------------------
// KNN: one 256-thread workgroup per point. dists in LDS, 20x parallel argmin.
// rank(j) = |xj|^2 - 2 xi.xj   (equivalent ordering to full distance)
// ---------------------------------------------------------------------------
template <int F>
__global__ __launch_bounds__(256)
void knn_kernel(const float* __restrict__ x, const float* __restrict__ sq,
                int* __restrict__ out) {
  __shared__ float sxi[F];
  __shared__ float sd[NPTS];
  __shared__ float sv[256];
  __shared__ int   si[256];
  const int tid = threadIdx.x;
  const int b = blockIdx.x >> 10;
  const int n = blockIdx.x & (NPTS - 1);
  const float* xb = x + (size_t)b * NPTS * F;
  for (int f = tid; f < F; f += 256) sxi[f] = xb[(size_t)n * F + f];
  __syncthreads();
  const float* sqb = sq + (size_t)b * NPTS;
  for (int j = tid; j < NPTS; j += 256) {
    const float* xj = xb + (size_t)j * F;
    float dot = 0.f;
#pragma unroll 4
    for (int f = 0; f < F; ++f) dot += sxi[f] * xj[f];
    sd[j] = sqb[j] - 2.f * dot;
  }
  __syncthreads();
  int* o = out + ((size_t)b * NPTS + n) * KNN_K;
  for (int k = 0; k < KNN_K; ++k) {
    float best = 3.4e38f; int bi = 1 << 30;
    for (int j = tid; j < NPTS; j += 256) {
      float v = sd[j];
      if (v < best || (v == best && j < bi)) { best = v; bi = j; }
    }
    sv[tid] = best; si[tid] = bi;
    __syncthreads();
    for (int s = 128; s > 0; s >>= 1) {
      if (tid < s) {
        float v2 = sv[tid + s]; int i2 = si[tid + s];
        if (v2 < sv[tid] || (v2 == sv[tid] && i2 < si[tid])) { sv[tid] = v2; si[tid] = i2; }
      }
      __syncthreads();
    }
    if (tid == 0) { o[k] = si[0]; sd[si[0]] = 3.5e38f; }
    __syncthreads();
  }
}

// ---------------------------------------------------------------------------
// fragment builders
// ---------------------------------------------------------------------------
__device__ inline v16h a_frag(const _Float16* p, int hiA) {
  // a[0..15] = p[2*hiA..2*hiA+15]  (contiguous k = 16h + s, matches swizzled B)
  v8h lo = *(const v8h*)(p + 2 * hiA);
  v8h hi = *(const v8h*)(p + 2 * hiA + 8);
  return __builtin_shufflevector(lo, hi, 0, 1, 2, 3, 4, 5, 6, 7,
                                 8, 9, 10, 11, 12, 13, 14, 15);
}
__device__ inline v16h b_frag(const _Float16* wp) {
  v8h lo = *(const v8h*)(wp);
  v8h hi = *(const v8h*)(wp + 8);
  return __builtin_shufflevector(lo, hi, 0, 1, 2, 3, 4, 5, 6, 7,
                                 8, 9, 10, 11, 12, 13, 14, 15);
}

// ---------------------------------------------------------------------------
// Fused DynamicEdgeConv MLP via WMMA f16->f32. One workgroup per point.
// B-matrix fragments read straight from pre-swizzled f16 weights in global
// (L2-resident), so the inner GEMM loops are barrier-free.
// ---------------------------------------------------------------------------
template <int F, int C>
__global__ __launch_bounds__(256)
void edgeconv_kernel(const float* __restrict__ x, const int* __restrict__ knn,
                     const _Float16* __restrict__ W1s, const float* __restrict__ b1v,
                     const float* __restrict__ g1v, const float* __restrict__ be1v,
                     const _Float16* __restrict__ W2s, const float* __restrict__ b2v,
                     const float* __restrict__ g2v, const float* __restrict__ be2v,
                     float* __restrict__ y) {
  constexpr int K2F = 2 * F;
  constexpr int KP1 = (K2F < 32) ? 32 : K2F;   // K padded for 16x16x32 WMMA
  constexpr int NT  = C / 16;                  // N-tiles
  constexpr int TPW = (2 * NT) / 8;            // wmma tile-tasks per wave
  static_assert((2 * NT) % 8 == 0, "task split");

  __shared__ _Float16 sE[32][KP1];
  __shared__ _Float16 sH[32][C];
  __shared__ unsigned smax[C];

  const int tid = threadIdx.x;
  // wave index is uniform per wave: tell the compiler so task/mt/nt scalarize
  const int wave = __builtin_amdgcn_readfirstlane(tid >> 5);
  const int lane = tid & 31;
  const int lane16 = lane & 15;
  const int hiA = (lane >> 4) * 8;    // acc row offset per lane half; A-frag K offset is 2*hiA
  const int hiB = (lane >> 4) * 16;   // B-frag K offset per lane half
  const int b = blockIdx.x >> 10;
  const int n = blockIdx.x & (NPTS - 1);

  for (int i = tid; i < 32 * KP1; i += 256) (&sE[0][0])[i] = (_Float16)0.0f;
  for (int i = tid; i < C; i += 256) smax[i] = 0u;
  __syncthreads();

  const float* xb = x + (size_t)b * NPTS * F;
  const float* xi = xb + (size_t)n * F;
  const int* id = knn + ((size_t)b * NPTS + n) * KNN_K;
  for (int i = tid; i < KNN_K * F; i += 256) {
    int r = i / F, f = i - r * F;
    int j = id[r];
    float vi = xi[f];
    float vj = xb[(size_t)j * F + f];
    sE[r][f]     = (_Float16)vi;            // x_i
    sE[r][F + f] = (_Float16)(vj - vi);     // x_j - x_i
  }
  __syncthreads();

  v8f acc[TPW];
#pragma unroll
  for (int t = 0; t < TPW; ++t)
#pragma unroll
    for (int r = 0; r < 8; ++r) acc[t][r] = 0.f;

  // ----- GEMM1: E[32,K2F] @ W1[K2F,C] ----- (no barriers inside)
  for (int kb = 0; kb < KP1; kb += 32) {
#pragma unroll
    for (int t = 0; t < TPW; ++t) {
      int task = wave * TPW + t;
      int mt = task & 1, nt = task >> 1;
      // mt folded into the DS address (scalar offset); duplicate loads across
      // tasks with the same mt are CSE'd by the compiler.
      v16h a = a_frag(&sE[mt * 16 + lane16][kb], hiA);
      const _Float16* wp = W1s + (((size_t)(kb >> 5) * C + nt * 16 + lane16) << 5) + hiB;
      if (t == 0 && kb + 32 < KP1) __builtin_prefetch(wp + (size_t)C * 32, 0, 3);
      acc[t] = __builtin_amdgcn_wmma_f32_16x16x32_f16(false, a, false, b_frag(wp),
                                                      (short)0, acc[t], false, false);
    }
  }

  // h1 = relu(bn1(. + b1)) -> sH (f16)
#pragma unroll
  for (int t = 0; t < TPW; ++t) {
    int task = wave * TPW + t;
    int mt = task & 1, nt = task >> 1;
    int c = nt * 16 + lane16;
    float gs = g1v[c] * BNS;
    float sh = b1v[c] * gs + be1v[c];
#pragma unroll
    for (int r = 0; r < 8; ++r) {
      float vv = acc[t][r] * gs + sh;
      vv = vv > 0.f ? vv : 0.f;
      sH[mt * 16 + r + hiA][c] = (_Float16)vv;
    }
  }
  __syncthreads();

#pragma unroll
  for (int t = 0; t < TPW; ++t)
#pragma unroll
    for (int r = 0; r < 8; ++r) acc[t][r] = 0.f;

  // ----- GEMM2: h1[32,C] @ W2[C,C] ----- (no barriers inside)
  for (int kb = 0; kb < C; kb += 32) {
#pragma unroll
    for (int t = 0; t < TPW; ++t) {
      int task = wave * TPW + t;
      int mt = task & 1, nt = task >> 1;
      v16h a = a_frag(&sH[mt * 16 + lane16][kb], hiA);
      const _Float16* wp = W2s + (((size_t)(kb >> 5) * C + nt * 16 + lane16) << 5) + hiB;
      if (t == 0 && kb + 32 < C) __builtin_prefetch(wp + (size_t)C * 32, 0, 3);
      acc[t] = __builtin_amdgcn_wmma_f32_16x16x32_f16(false, a, false, b_frag(wp),
                                                      (short)0, acc[t], false, false);
    }
  }

  // epilogue: bn+relu, masked max over the 20 valid rows
#pragma unroll
  for (int t = 0; t < TPW; ++t) {
    int task = wave * TPW + t;
    int mt = task & 1, nt = task >> 1;
    int c = nt * 16 + lane16;
    float gs = g2v[c] * BNS;
    float sh = b2v[c] * gs + be2v[c];
    float m = 0.f;                                // relu => identity 0 is safe
#pragma unroll
    for (int r = 0; r < 8; ++r) {
      float vv = acc[t][r] * gs + sh;
      vv = vv > 0.f ? vv : 0.f;
      int row = mt * 16 + r + hiA;
      if (row < KNN_K) m = m > vv ? m : vv;
    }
    float mo = __shfl_xor(m, 16, 32);
    m = m > mo ? m : mo;
    if (lane < 16) atomicMax(&smax[c], __float_as_uint(m));
  }
  __syncthreads();
  float* yo = y + ((size_t)b * NPTS + n) * C;
  for (int c2 = tid; c2 < C; c2 += 256) yo[c2] = __uint_as_float(smax[c2]);
}

// ---------------------------------------------------------------------------
// Tiled WMMA GEMM for FiLM: Y[8192,C] = X[8192,C] @ W[C,C] + bias
// 64x64 tile per block; A staged f16 in LDS (b128 in/out), B direct from
// pre-swizzled f16 weights in global.
// ---------------------------------------------------------------------------
template <int C>
__global__ __launch_bounds__(256)
void gemm_film(const float* __restrict__ X, const _Float16* __restrict__ Wsw,
               const float* __restrict__ bias, float* __restrict__ Y) {
  static_assert(C % 64 == 0, "tile");
  __shared__ _Float16 sA[64][32];
  const int tid = threadIdx.x;
  const int wave = __builtin_amdgcn_readfirstlane(tid >> 5);
  const int lane = tid & 31;
  const int lane16 = lane & 15;
  const int hiA = (lane >> 4) * 8;
  const int hiB = (lane >> 4) * 16;
  const int m0 = blockIdx.x * 64;
  const int n0 = blockIdx.y * 64;

  v8f acc[2];
#pragma unroll
  for (int t = 0; t < 2; ++t)
#pragma unroll
    for (int r = 0; r < 8; ++r) acc[t][r] = 0.f;

  for (int kb = 0; kb < C; kb += 32) {
    __syncthreads();
    {  // stage A block: 64x32, 8 halves per thread (vector in/out)
      int r = tid >> 2, c = (tid & 3) * 8;
      const float* src = &X[(size_t)(m0 + r) * C + kb + c];
      v8h h;
#pragma unroll
      for (int j = 0; j < 8; ++j) h[j] = (_Float16)src[j];
      *(v8h*)&sA[r][c] = h;
    }
    __syncthreads();
#pragma unroll
    for (int t = 0; t < 2; ++t) {
      int task = wave * 2 + t;
      int mt = task & 3, nt = task >> 2;
      v16h a = a_frag(&sA[mt * 16 + lane16][0], hiA);
      const _Float16* wp = Wsw + (((size_t)(kb >> 5) * C + n0 + nt * 16 + lane16) << 5) + hiB;
      if (t == 0 && kb + 32 < C) __builtin_prefetch(wp + (size_t)C * 32, 0, 3);
      acc[t] = __builtin_amdgcn_wmma_f32_16x16x32_f16(false, a, false, b_frag(wp),
                                                      (short)0, acc[t], false, false);
    }
  }
#pragma unroll
  for (int t = 0; t < 2; ++t) {
    int task = wave * 2 + t;
    int mt = task & 3, nt = task >> 2;
    int col = n0 + nt * 16 + lane16;
    float bi = bias[col];
#pragma unroll
    for (int r = 0; r < 8; ++r) {
      int row = m0 + mt * 16 + r + hiA;
      Y[(size_t)row * C + col] = acc[t][r] + bi;
    }
  }
}

// y = gate * x + shift (elementwise FiLM combine)
__global__ void film_combine_kernel(const float* __restrict__ xc,
                                    const float* __restrict__ gate,
                                    const float* __restrict__ shf,
                                    float* __restrict__ y, int total) {
  int i = blockIdx.x * blockDim.x + threadIdx.x;
  if (i < total) y[i] = gate[i] * xc[i] + shf[i];
}

// mean over N points -> pooled[b, colOff + c]
__global__ void meanpool_kernel(const float* __restrict__ x, int C,
                                float* __restrict__ pooled, int colOff) {
  int i = blockIdx.x * blockDim.x + threadIdx.x;
  if (i >= BATCH * C) return;
  int b = i / C, c = i - b * C;
  const float* p = x + ((size_t)b * NPTS) * C + c;
  float s = 0.f;
  for (int n = 0; n < NPTS; ++n) s += p[(size_t)n * C];
  pooled[(size_t)b * 960 + colOff + c] = s * (1.0f / NPTS);
}

// small dense head: Y = f(X @ W + b), optional bn+relu (mode=1)
__global__ void fc_kernel(const float* __restrict__ X, const float* __restrict__ W,
                          const float* __restrict__ bv, const float* __restrict__ gv,
                          const float* __restrict__ bev, float* __restrict__ Y,
                          int M, int Kd, int N, int mode) {
  int i = blockIdx.x * blockDim.x + threadIdx.x;
  if (i >= M * N) return;
  int m = i / N, n = i - m * N;
  float acc = 0.f;
  for (int k = 0; k < Kd; ++k) acc += X[(size_t)m * Kd + k] * W[(size_t)k * N + n];
  float v = acc + bv[n];
  if (mode) { v = v * (gv[n] * BNS) + bev[n]; v = v > 0.f ? v : 0.f; }
  Y[i] = v;
}

// ---------------------------------------------------------------------------
extern "C" void kernel_launch(void* const* d_in, const int* in_sizes, int n_in,
                              void* d_out, int out_size, void* d_ws, size_t ws_size,
                              hipStream_t stream) {
  (void)in_sizes; (void)n_in; (void)out_size; (void)ws_size;
  const float* pos = (const float*)d_in[0];
  const float* nrm = (const float*)d_in[1];
  auto P = [&](int i) { return (const float*)d_in[i]; };

  char* ws = (char*)d_ws;
  size_t off = 0;
  auto alloc = [&](size_t bytes) -> char* {
    char* p = ws + off;
    off = (off + bytes + 255) & ~(size_t)255;
    return p;
  };
  float* x0 = (float*)alloc((size_t)BATCH * NPTS * 6 * 4);
  float* x1 = (float*)alloc((size_t)BATCH * NPTS * 64 * 4);
  float* x2 = (float*)alloc((size_t)BATCH * NPTS * 128 * 4);
  float* x3 = (float*)alloc((size_t)BATCH * NPTS * 256 * 4);
  float* x4 = (float*)alloc((size_t)BATCH * NPTS * 512 * 4);
  float* cv = (float*)alloc((size_t)BATCH * NPTS * 512 * 4);
  float* gt = (float*)alloc((size_t)BATCH * NPTS * 512 * 4);
  float* sf = (float*)alloc((size_t)BATCH * NPTS * 512 * 4);
  float* sq = (float*)alloc((size_t)BATCH * NPTS * 4);
  int*   kn = (int*)alloc((size_t)BATCH * NPTS * KNN_K * 4);
  float* pooled = (float*)alloc((size_t)BATCH * 960 * 4);
  float* h1 = (float*)alloc((size_t)BATCH * 512 * 4);
  float* h2 = (float*)alloc((size_t)BATCH * 256 * 4);
  // f16 swizzled weights
  _Float16* c1w1 = (_Float16*)alloc((size_t)32 * 64 * 2);
  _Float16* c1w2 = (_Float16*)alloc((size_t)64 * 64 * 2);
  _Float16* c2w1 = (_Float16*)alloc((size_t)128 * 128 * 2);
  _Float16* c2w2 = (_Float16*)alloc((size_t)128 * 128 * 2);
  _Float16* c3w1 = (_Float16*)alloc((size_t)256 * 256 * 2);
  _Float16* c3w2 = (_Float16*)alloc((size_t)256 * 256 * 2);
  _Float16* c4w1 = (_Float16*)alloc((size_t)512 * 512 * 2);
  _Float16* c4w2 = (_Float16*)alloc((size_t)512 * 512 * 2);
  _Float16* f1g = (_Float16*)alloc((size_t)64 * 64 * 2);
  _Float16* f1b = (_Float16*)alloc((size_t)64 * 64 * 2);
  _Float16* f2g = (_Float16*)alloc((size_t)128 * 128 * 2);
  _Float16* f2b = (_Float16*)alloc((size_t)128 * 128 * 2);
  _Float16* f3g = (_Float16*)alloc((size_t)256 * 256 * 2);
  _Float16* f3b = (_Float16*)alloc((size_t)256 * 256 * 2);
  _Float16* f4g = (_Float16*)alloc((size_t)512 * 512 * 2);
  _Float16* f4b = (_Float16*)alloc((size_t)512 * 512 * 2);

  const dim3 blk(256);
  const int BN = BATCH * NPTS;  // 8192
  auto gsz = [](int total) { return dim3((total + 255) / 256); };

  // ---- weight swizzle passes (tiny; L2-resident afterwards) ----
  wswizzle_kernel<<<gsz(32 * 64), blk, 0, stream>>>(P(2), c1w1, 12, 32, 64);
  wswizzle_kernel<<<gsz(64 * 64), blk, 0, stream>>>(P(6), c1w2, 64, 64, 64);
  wswizzle_kernel<<<gsz(128 * 128), blk, 0, stream>>>(P(14), c2w1, 128, 128, 128);
  wswizzle_kernel<<<gsz(128 * 128), blk, 0, stream>>>(P(18), c2w2, 128, 128, 128);
  wswizzle_kernel<<<gsz(256 * 256), blk, 0, stream>>>(P(26), c3w1, 256, 256, 256);
  wswizzle_kernel<<<gsz(256 * 256), blk, 0, stream>>>(P(30), c3w2, 256, 256, 256);
  wswizzle_kernel<<<gsz(512 * 512), blk, 0, stream>>>(P(38), c4w1, 512, 512, 512);
  wswizzle_kernel<<<gsz(512 * 512), blk, 0, stream>>>(P(42), c4w2, 512, 512, 512);
  wswizzle_kernel<<<gsz(64 * 64), blk, 0, stream>>>(P(10), f1g, 64, 64, 64);
  wswizzle_kernel<<<gsz(64 * 64), blk, 0, stream>>>(P(12), f1b, 64, 64, 64);
  wswizzle_kernel<<<gsz(128 * 128), blk, 0, stream>>>(P(22), f2g, 128, 128, 128);
  wswizzle_kernel<<<gsz(128 * 128), blk, 0, stream>>>(P(24), f2b, 128, 128, 128);
  wswizzle_kernel<<<gsz(256 * 256), blk, 0, stream>>>(P(34), f3g, 256, 256, 256);
  wswizzle_kernel<<<gsz(256 * 256), blk, 0, stream>>>(P(36), f3b, 256, 256, 256);
  wswizzle_kernel<<<gsz(512 * 512), blk, 0, stream>>>(P(46), f4g, 512, 512, 512);
  wswizzle_kernel<<<gsz(512 * 512), blk, 0, stream>>>(P(48), f4b, 512, 512, 512);

  concat_kernel<<<gsz(BN * 6), blk, 0, stream>>>(pos, nrm, x0);

  // ---- layer 1: F=6, C=64 ----
  sqnorm_kernel<6><<<gsz(BN), blk, 0, stream>>>(x0, sq);
  knn_kernel<6><<<dim3(BN), blk, 0, stream>>>(x0, sq, kn);
  edgeconv_kernel<6, 64><<<dim3(BN), blk, 0, stream>>>(
      x0, kn, c1w1, P(3), P(4), P(5), c1w2, P(7), P(8), P(9), cv);
  gemm_film<64><<<dim3(BN / 64, 1), blk, 0, stream>>>(cv, f1g, P(11), gt);
  gemm_film<64><<<dim3(BN / 64, 1), blk, 0, stream>>>(cv, f1b, P(13), sf);
  film_combine_kernel<<<gsz(BN * 64), blk, 0, stream>>>(cv, gt, sf, x1, BN * 64);

  // ---- layer 2: F=64, C=128 ----
  sqnorm_kernel<64><<<gsz(BN), blk, 0, stream>>>(x1, sq);
  knn_kernel<64><<<dim3(BN), blk, 0, stream>>>(x1, sq, kn);
  edgeconv_kernel<64, 128><<<dim3(BN), blk, 0, stream>>>(
      x1, kn, c2w1, P(15), P(16), P(17), c2w2, P(19), P(20), P(21), cv);
  gemm_film<128><<<dim3(BN / 64, 2), blk, 0, stream>>>(cv, f2g, P(23), gt);
  gemm_film<128><<<dim3(BN / 64, 2), blk, 0, stream>>>(cv, f2b, P(25), sf);
  film_combine_kernel<<<gsz(BN * 128), blk, 0, stream>>>(cv, gt, sf, x2, BN * 128);

  // ---- layer 3: F=128, C=256 ----
  sqnorm_kernel<128><<<gsz(BN), blk, 0, stream>>>(x2, sq);
  knn_kernel<128><<<dim3(BN), blk, 0, stream>>>(x2, sq, kn);
  edgeconv_kernel<128, 256><<<dim3(BN), blk, 0, stream>>>(
      x2, kn, c3w1, P(27), P(28), P(29), c3w2, P(31), P(32), P(33), cv);
  gemm_film<256><<<dim3(BN / 64, 4), blk, 0, stream>>>(cv, f3g, P(35), gt);
  gemm_film<256><<<dim3(BN / 64, 4), blk, 0, stream>>>(cv, f3b, P(37), sf);
  film_combine_kernel<<<gsz(BN * 256), blk, 0, stream>>>(cv, gt, sf, x3, BN * 256);

  // ---- layer 4: F=256, C=512 ----
  sqnorm_kernel<256><<<gsz(BN), blk, 0, stream>>>(x3, sq);
  knn_kernel<256><<<dim3(BN), blk, 0, stream>>>(x3, sq, kn);
  edgeconv_kernel<256, 512><<<dim3(BN), blk, 0, stream>>>(
      x3, kn, c4w1, P(39), P(40), P(41), c4w2, P(43), P(44), P(45), cv);
  gemm_film<512><<<dim3(BN / 64, 8), blk, 0, stream>>>(cv, f4g, P(47), gt);
  gemm_film<512><<<dim3(BN / 64, 8), blk, 0, stream>>>(cv, f4b, P(49), sf);
  film_combine_kernel<<<gsz(BN * 512), blk, 0, stream>>>(cv, gt, sf, x4, BN * 512);

  // ---- pooling: concat of per-layer means -> [8, 960] ----
  meanpool_kernel<<<gsz(BATCH * 64), blk, 0, stream>>>(x1, 64, pooled, 0);
  meanpool_kernel<<<gsz(BATCH * 128), blk, 0, stream>>>(x2, 128, pooled, 64);
  meanpool_kernel<<<gsz(BATCH * 256), blk, 0, stream>>>(x3, 256, pooled, 192);
  meanpool_kernel<<<gsz(BATCH * 512), blk, 0, stream>>>(x4, 512, pooled, 448);

  // ---- head ----
  fc_kernel<<<gsz(BATCH * 512), blk, 0, stream>>>(
      pooled, P(50), P(51), P(52), P(53), h1, BATCH, 960, 512, 1);
  fc_kernel<<<gsz(BATCH * 256), blk, 0, stream>>>(
      h1, P(54), P(55), P(56), P(57), h2, BATCH, 512, 256, 1);
  fc_kernel<<<dim3(1), dim3(32), 0, stream>>>(
      h2, P(58), P(59), P(59), P(59), (float*)d_out, BATCH, 256, 1, 0);
}
